// FinancialMultiModalFusion_55808805044797
// MI455X (gfx1250) — matrix operand order repaired and checked by reading, not compiled, not run on hardware
//
#include <hip/hip_runtime.h>
#include <math.h>

// ---------------------------------------------------------------------------
// FinancialMultiModalFusion for MI455X (gfx1250, wave32, WMMA f32_16x16x32_f16)
// ---------------------------------------------------------------------------
typedef _Float16 half_t;
typedef __attribute__((ext_vector_type(16))) _Float16 v16h;
typedef __attribute__((ext_vector_type(8)))  _Float16 v8h;
typedef __attribute__((ext_vector_type(4)))  _Float16 v4h;
typedef __attribute__((ext_vector_type(4)))  float    v4f;
typedef __attribute__((ext_vector_type(8)))  float    v8f;

#define BATCH 16384
#define HDIM  512
#define BH    ((size_t)BATCH * HDIM)

__device__ const int g_src[12] = {1,2,3, 0,2,3, 0,1,3, 0,1,2};
__device__ const int g_cat[4]  = {2,3,1,0};

__device__ __forceinline__ v8f wmma_f16(v16h a, v16h b, v8f c) {
  return __builtin_amdgcn_wmma_f32_16x16x32_f16(
      /*neg_a=*/false, a, /*neg_b=*/false, b,
      /*c_mod=*/(short)0, c, /*reuse_a=*/false, /*reuse_b=*/false);
}

// A fragment from LDS tile As[rows][32] (row-major, f16).
__device__ __forceinline__ v16h frag_a(const half_t* As, int row0, int lane) {
  int row = row0 + (lane & 15);
  int k0  = (lane >> 4) * 8;
  const half_t* p = As + row * 32;
  union { v16h v; v8h h[2]; } u;
  u.h[0] = *(const v8h*)(p + k0);
  u.h[1] = *(const v8h*)(p + k0 + 16);
  return u.v;
}

// B fragment from LDS tile Bs[cols][32] (K-contiguous per column, f16).
__device__ __forceinline__ v16h frag_b(const half_t* Bs, int col0, int lane) {
  int col = col0 + (lane & 15);
  int k0  = (lane >> 4) * 8;
  const half_t* p = Bs + col * 32;
  union { v16h v; v8h h[2]; } u;
  u.h[0] = *(const v8h*)(p + k0);
  u.h[1] = *(const v8h*)(p + k0 + 16);
  return u.v;
}

// ---------------------------------------------------------------------------
// Vectorized tile staging: tiles are ROWS x 32 (f16 in LDS).
// f16 source: 16B chunks (v8h).  f32 source: float4 load -> half4 store.
// ---------------------------------------------------------------------------
template <int ROWS>
__device__ __forceinline__ void stage_f16(const half_t* __restrict__ g, size_t stride,
                                          int rb, int kb, half_t* __restrict__ s, int tid) {
  #pragma unroll
  for (int i = 0; i < (ROWS * 4) / 256; ++i) {
    int e = tid + i * 256;
    int r = e >> 2, o = (e & 3) * 8;
    *(v8h*)(s + r * 32 + o) = *(const v8h*)(g + (size_t)(rb + r) * stride + kb + o);
  }
}

template <int ROWS>
__device__ __forceinline__ void stage_f32(const float* __restrict__ g, size_t stride,
                                          int rb, int kb, half_t* __restrict__ s, int tid) {
  #pragma unroll
  for (int i = 0; i < (ROWS * 8) / 256; ++i) {
    int e = tid + i * 256;
    int r = e >> 3, o = (e & 7) * 4;
    v4f x = *(const v4f*)(g + (size_t)(rb + r) * stride + kb + o);
    *(v4h*)(s + r * 32 + o) = __builtin_convertvector(x, v4h);
  }
}

// ---------------------------------------------------------------------------
// Prep kernels
// ---------------------------------------------------------------------------
__global__ void k_transpose_f16(const float* __restrict__ W, half_t* __restrict__ WT,
                                int K, int N, long total) {
  long o = (long)blockIdx.x * 256 + threadIdx.x;
  if (o >= total) return;
  long per = (long)K * N;
  long b   = o / per;
  long rem = o - b * per;
  int  n   = (int)(rem / K);
  int  k   = (int)(rem - (long)n * K);
  WT[o] = (half_t)W[b * per + (long)k * N + n];
}

__global__ void k_convert_f16(const float* __restrict__ W, half_t* __restrict__ O, long total) {
  long o = (long)blockIdx.x * 256 + threadIdx.x;
  if (o < total) O[o] = (half_t)W[o];
}

// bsum[m][n] = sum_{j=0..2, p=3m+j} ( bv[p] @ Wo[p] + bo[p] )[n]
__global__ void k_bsum(const float* __restrict__ bv, const float* __restrict__ Wo,
                       const float* __restrict__ bo, float* __restrict__ bsum) {
  int idx = blockIdx.x * 256 + threadIdx.x;
  if (idx >= 4 * HDIM) return;
  int m = idx >> 9, n = idx & 511;
  float s = 0.f;
  for (int j = 0; j < 3; ++j) {
    int p = 3 * m + j;
    const float* bvp = bv + p * HDIM;
    const float* Wop = Wo + (size_t)p * HDIM * HDIM;
    float t = 0.f;
    for (int k = 0; k < HDIM; ++k) t += bvp[k] * Wop[(size_t)k * HDIM + n];
    s += t + bo[p * HDIM + n];
  }
  bsum[idx] = s;
}

// ---------------------------------------------------------------------------
// Batched f16 GEMM (M=N=K=512): WeffT[z][n][k] = (Wv[z][k]@Wo[z])[n].
// ---------------------------------------------------------------------------
__global__ __launch_bounds__(256)
void k_gemm_ff16(const half_t* __restrict__ A, const half_t* __restrict__ Bt,
                 half_t* __restrict__ C) {
  __shared__ __align__(64) half_t As[128 * 32];
  __shared__ __align__(64) half_t Bs[128 * 32];
  int z = blockIdx.z;
  const half_t* Ab = A  + (size_t)z * HDIM * HDIM;
  const half_t* Bb = Bt + (size_t)z * HDIM * HDIM;
  half_t*       Cb = C  + (size_t)z * HDIM * HDIM;
  int rb = blockIdx.x * 128, cb = blockIdx.y * 128;
  int tid = threadIdx.x, lane = tid & 31, wid = tid >> 5;
  int wm = wid >> 2, wn = wid & 3;            // 2x4 waves -> wave tile 64x32
  v8f acc[4][2];
  #pragma unroll
  for (int i = 0; i < 4; ++i)
    #pragma unroll
    for (int j = 0; j < 2; ++j) acc[i][j] = (v8f){0,0,0,0,0,0,0,0};

  for (int kb = 0; kb < HDIM; kb += 32) {
    stage_f16<128>(Ab, HDIM, rb, kb, As, tid);
    stage_f16<128>(Bb, HDIM, cb, kb, Bs, tid);
    __syncthreads();
    v16h bf[2];
    #pragma unroll
    for (int nt = 0; nt < 2; ++nt) bf[nt] = frag_b(Bs, wn * 32 + nt * 16, lane);
    #pragma unroll
    for (int mt = 0; mt < 4; ++mt) {
      v16h af = frag_a(As, wm * 64 + mt * 16, lane);
      #pragma unroll
      for (int nt = 0; nt < 2; ++nt) acc[mt][nt] = wmma_f16(af, bf[nt], acc[mt][nt]);
    }
    __syncthreads();
  }
  int colHalf = lane & 15, rowHalf = lane >> 4;
  #pragma unroll
  for (int mt = 0; mt < 4; ++mt)
    #pragma unroll
    for (int nt = 0; nt < 2; ++nt) {
      int col = cb + wn * 32 + nt * 16 + colHalf;
      int row0 = rb + wm * 64 + mt * 16 + rowHalf * 8;
      #pragma unroll
      for (int r = 0; r < 8; ++r)
        Cb[(size_t)(row0 + r) * HDIM + col] = (half_t)acc[mt][nt][r];
    }
}

// ---------------------------------------------------------------------------
// GEMM (M x K) x (K x 512) + bias -> LayerNorm -> (opt ReLU). BM=64, BN=512.
// ---------------------------------------------------------------------------
template <typename TA, bool RELU>
__global__ __launch_bounds__(256)
void k_gemm_ln(const TA* __restrict__ X, int K,
               const half_t* __restrict__ WT,   // [512][K] f16 (col-major of W)
               const float* __restrict__ bias,
               const float* __restrict__ gamma,
               const float* __restrict__ beta,
               half_t* __restrict__ outH,       // nullable
               float* __restrict__ outF) {      // nullable
  __shared__ __align__(64) half_t As[64 * 32];
  __shared__ __align__(64) half_t Bs[512 * 32];
  __shared__ float rsum[64], rsq[64], rmean[64], rrstd[64];

  int tid = threadIdx.x, lane = tid & 31, wid = tid >> 5;
  int wm = wid >> 2, wn = wid & 3;              // 2x4 waves -> wave tile 32x128
  int rowBlock = blockIdx.x * 64;

  v8f acc[2][8];
  #pragma unroll
  for (int i = 0; i < 2; ++i)
    #pragma unroll
    for (int j = 0; j < 8; ++j) acc[i][j] = (v8f){0,0,0,0,0,0,0,0};

  for (int kb = 0; kb < K; kb += 32) {
    if constexpr (sizeof(TA) == 4)
      stage_f32<64>((const float*)X, K, rowBlock, kb, As, tid);
    else
      stage_f16<64>((const half_t*)X, K, rowBlock, kb, As, tid);
    stage_f16<512>(WT, K, 0, kb, Bs, tid);
    __syncthreads();
    v16h bf[8];
    #pragma unroll
    for (int nt = 0; nt < 8; ++nt) bf[nt] = frag_b(Bs, wn * 128 + nt * 16, lane);
    #pragma unroll
    for (int mt = 0; mt < 2; ++mt) {
      v16h af = frag_a(As, wm * 32 + mt * 16, lane);
      #pragma unroll
      for (int nt = 0; nt < 8; ++nt) acc[mt][nt] = wmma_f16(af, bf[nt], acc[mt][nt]);
    }
    __syncthreads();
  }

  if (tid < 64) { rsum[tid] = 0.f; rsq[tid] = 0.f; }
  __syncthreads();

  int colHalf = lane & 15, rowHalf = lane >> 4;
  #pragma unroll
  for (int mt = 0; mt < 2; ++mt) {
    #pragma unroll
    for (int r = 0; r < 8; ++r) {
      float s = 0.f, q = 0.f;
      #pragma unroll
      for (int nt = 0; nt < 8; ++nt) {
        int col = wn * 128 + nt * 16 + colHalf;
        float v = acc[mt][nt][r] + bias[col];
        acc[mt][nt][r] = v;
        s += v; q += v * v;
      }
      #pragma unroll
      for (int m = 1; m < 16; m <<= 1) {
        s += __shfl_xor(s, m, 32);
        q += __shfl_xor(q, m, 32);
      }
      if (colHalf == 0) {
        int row = wm * 32 + mt * 16 + rowHalf * 8 + r;
        atomicAdd(&rsum[row], s);
        atomicAdd(&rsq[row], q);
      }
    }
  }
  __syncthreads();
  if (tid < 64) {
    float m = rsum[tid] * (1.f / 512.f);
    float v = rsq[tid] * (1.f / 512.f) - m * m;
    rmean[tid] = m;
    rrstd[tid] = rsqrtf(v + 1e-5f);
  }
  __syncthreads();

  #pragma unroll
  for (int mt = 0; mt < 2; ++mt) {
    int row0 = wm * 32 + mt * 16 + rowHalf * 8;
    #pragma unroll
    for (int r = 0; r < 8; ++r) {
      int row = row0 + r;
      float mu = rmean[row], rs = rrstd[row];
      size_t gro = (size_t)(rowBlock + row) * HDIM;
      #pragma unroll
      for (int nt = 0; nt < 8; ++nt) {
        int col = wn * 128 + nt * 16 + colHalf;
        float v = (acc[mt][nt][r] - mu) * rs * gamma[col] + beta[col];
        if (RELU) v = fmaxf(v, 0.f);
        if (outH) outH[gro + col] = (half_t)v;
        if (outF) outF[gro + col] = v;
      }
    }
  }
}

// ---------------------------------------------------------------------------
// Attention collapse: att[m] = P[m] + bsum[m] + sum_{seg} P[src]@Weff[p].
// K=1536 GEMM with per-512-segment A/B remap. BM=BN=128. z = query modality.
// ---------------------------------------------------------------------------
__global__ __launch_bounds__(256)
void k_attn(const half_t* __restrict__ P16, const half_t* __restrict__ WeffT,
            const float* __restrict__ bsum,
            float* __restrict__ attF, half_t* __restrict__ attH) {
  __shared__ __align__(64) half_t As[128 * 32];
  __shared__ __align__(64) half_t Bs[128 * 32];
  int m = blockIdx.z;
  int rb = blockIdx.x * 128, cb = blockIdx.y * 128;
  int tid = threadIdx.x, lane = tid & 31, wid = tid >> 5;
  int wm = wid >> 2, wn = wid & 3;
  v8f acc[4][2];
  #pragma unroll
  for (int i = 0; i < 4; ++i)
    #pragma unroll
    for (int j = 0; j < 2; ++j) acc[i][j] = (v8f){0,0,0,0,0,0,0,0};

  for (int seg = 0; seg < 3; ++seg) {
    int p = 3 * m + seg;
    const half_t* Ab = P16 + (size_t)g_src[p] * BH;
    const half_t* Bb = WeffT + (size_t)p * HDIM * HDIM;
    for (int kb = 0; kb < HDIM; kb += 32) {
      stage_f16<128>(Ab, HDIM, rb, kb, As, tid);
      stage_f16<128>(Bb, HDIM, cb, kb, Bs, tid);
      __syncthreads();
      v16h bf[2];
      #pragma unroll
      for (int nt = 0; nt < 2; ++nt) bf[nt] = frag_b(Bs, wn * 32 + nt * 16, lane);
      #pragma unroll
      for (int mt = 0; mt < 4; ++mt) {
        v16h af = frag_a(As, wm * 64 + mt * 16, lane);
        #pragma unroll
        for (int nt = 0; nt < 2; ++nt) acc[mt][nt] = wmma_f16(af, bf[nt], acc[mt][nt]);
      }
      __syncthreads();
    }
  }
  int colHalf = lane & 15, rowHalf = lane >> 4;
  #pragma unroll
  for (int mt = 0; mt < 4; ++mt)
    #pragma unroll
    for (int nt = 0; nt < 2; ++nt) {
      int gcol = cb + wn * 32 + nt * 16 + colHalf;
      int row0 = rb + wm * 64 + mt * 16 + rowHalf * 8;
      float bs = bsum[m * HDIM + gcol];
      #pragma unroll
      for (int r = 0; r < 8; ++r) {
        size_t o = (size_t)m * BH + (size_t)(row0 + r) * HDIM + gcol;
        float v = acc[mt][nt][r] + bs + (float)P16[o];
        attF[o] = v;
        attH[o] = (half_t)v;
      }
    }
}

// ---------------------------------------------------------------------------
// hidden = relu(cat @ Wi1 + bi1) : K=2048 with cat = att[{2,3,1,0}] remap.
// ---------------------------------------------------------------------------
__global__ __launch_bounds__(256)
void k_cat_relu(const half_t* __restrict__ att16, const half_t* __restrict__ WT,
                const float* __restrict__ bias, half_t* __restrict__ outH) {
  __shared__ __align__(64) half_t As[128 * 32];
  __shared__ __align__(64) half_t Bs[128 * 32];
  int rb = blockIdx.x * 128, cb = blockIdx.y * 128;
  int tid = threadIdx.x, lane = tid & 31, wid = tid >> 5;
  int wm = wid >> 2, wn = wid & 3;
  v8f acc[4][2];
  #pragma unroll
  for (int i = 0; i < 4; ++i)
    #pragma unroll
    for (int j = 0; j < 2; ++j) acc[i][j] = (v8f){0,0,0,0,0,0,0,0};

  for (int seg = 0; seg < 4; ++seg) {
    const half_t* Ab = att16 + (size_t)g_cat[seg] * BH;
    for (int kb = 0; kb < HDIM; kb += 32) {
      stage_f16<128>(Ab, HDIM, rb, kb, As, tid);
      stage_f16<128>(WT, 2048, cb, seg * HDIM + kb, Bs, tid);
      __syncthreads();
      v16h bf[2];
      #pragma unroll
      for (int nt = 0; nt < 2; ++nt) bf[nt] = frag_b(Bs, wn * 32 + nt * 16, lane);
      #pragma unroll
      for (int mt = 0; mt < 4; ++mt) {
        v16h af = frag_a(As, wm * 64 + mt * 16, lane);
        #pragma unroll
        for (int nt = 0; nt < 2; ++nt) acc[mt][nt] = wmma_f16(af, bf[nt], acc[mt][nt]);
      }
      __syncthreads();
    }
  }
  int colHalf = lane & 15, rowHalf = lane >> 4;
  #pragma unroll
  for (int mt = 0; mt < 4; ++mt)
    #pragma unroll
    for (int nt = 0; nt < 2; ++nt) {
      int gcol = cb + wn * 32 + nt * 16 + colHalf;
      int row0 = rb + wm * 64 + mt * 16 + rowHalf * 8;
      float b = bias[gcol];
      #pragma unroll
      for (int r = 0; r < 8; ++r)
        outH[(size_t)(row0 + r) * HDIM + gcol] = (half_t)fmaxf(acc[mt][nt][r] + b, 0.f);
    }
}

// ---------------------------------------------------------------------------
// fused = sigmoid(cat@Wg + bg) * tanh(cat@Wc + bc) : dual-accumulator GEMM.
// BM=128, BN=64.
// ---------------------------------------------------------------------------
__global__ __launch_bounds__(256)
void k_cat_gate(const half_t* __restrict__ att16,
                const half_t* __restrict__ WgT, const half_t* __restrict__ WcT,
                const float* __restrict__ bg, const float* __restrict__ bc,
                half_t* __restrict__ fused16) {
  __shared__ __align__(64) half_t As[128 * 32];
  __shared__ __align__(64) half_t Bg[64 * 32];
  __shared__ __align__(64) half_t Bc[64 * 32];
  int rb = blockIdx.x * 128, cb = blockIdx.y * 64;
  int tid = threadIdx.x, lane = tid & 31, wid = tid >> 5;
  int wm = wid >> 1, wn = wid & 1;              // 4x2 waves -> wave tile 32x32
  v8f ag[2][2], ac[2][2];
  #pragma unroll
  for (int i = 0; i < 2; ++i)
    #pragma unroll
    for (int j = 0; j < 2; ++j) { ag[i][j] = (v8f){0,0,0,0,0,0,0,0}; ac[i][j] = ag[i][j]; }

  for (int seg = 0; seg < 4; ++seg) {
    const half_t* Ab = att16 + (size_t)g_cat[seg] * BH;
    for (int kb = 0; kb < HDIM; kb += 32) {
      stage_f16<128>(Ab, HDIM, rb, kb, As, tid);
      stage_f16<64>(WgT, 2048, cb, seg * HDIM + kb, Bg, tid);
      stage_f16<64>(WcT, 2048, cb, seg * HDIM + kb, Bc, tid);
      __syncthreads();
      v16h bgf[2], bcf[2];
      #pragma unroll
      for (int nt = 0; nt < 2; ++nt) {
        bgf[nt] = frag_b(Bg, wn * 32 + nt * 16, lane);
        bcf[nt] = frag_b(Bc, wn * 32 + nt * 16, lane);
      }
      #pragma unroll
      for (int mt = 0; mt < 2; ++mt) {
        v16h af = frag_a(As, wm * 32 + mt * 16, lane);
        #pragma unroll
        for (int nt = 0; nt < 2; ++nt) {
          ag[mt][nt] = wmma_f16(af, bgf[nt], ag[mt][nt]);
          ac[mt][nt] = wmma_f16(af, bcf[nt], ac[mt][nt]);
        }
      }
      __syncthreads();
    }
  }
  int colHalf = lane & 15, rowHalf = lane >> 4;
  #pragma unroll
  for (int mt = 0; mt < 2; ++mt)
    #pragma unroll
    for (int nt = 0; nt < 2; ++nt) {
      int gcol = cb + wn * 32 + nt * 16 + colHalf;
      int row0 = rb + wm * 32 + mt * 16 + rowHalf * 8;
      float vbg = bg[gcol], vbc = bc[gcol];
      #pragma unroll
      for (int r = 0; r < 8; ++r) {
        float g = ag[mt][nt][r] + vbg;
        float c = ac[mt][nt][r] + vbc;
        float f = (1.f / (1.f + expf(-g))) * tanhf(c);
        fused16[(size_t)(row0 + r) * HDIM + gcol] = (half_t)f;
      }
    }
}

// ---------------------------------------------------------------------------
// importance = softmax(hidden @ Wi2 + bi2) over 4. One wave per row.
// ---------------------------------------------------------------------------
__global__ __launch_bounds__(256)
void k_importance(const half_t* __restrict__ hidden16, const float* __restrict__ Wi2,
                  const float* __restrict__ bi2, float* __restrict__ imp) {
  int row = (int)((blockIdx.x * 256 + threadIdx.x) >> 5);
  int lane = threadIdx.x & 31;
  if (row >= BATCH) return;
  const half_t* h = hidden16 + (size_t)row * HDIM;
  float p0 = 0, p1 = 0, p2 = 0, p3 = 0;
  #pragma unroll
  for (int i = 0; i < 2; ++i) {
    v8h hv8 = *(const v8h*)(h + lane * 16 + i * 8);
    #pragma unroll
    for (int j = 0; j < 8; ++j) {
      int k = lane * 16 + i * 8 + j;
      float hv = (float)hv8[j];
      p0 += hv * Wi2[k * 4 + 0];
      p1 += hv * Wi2[k * 4 + 1];
      p2 += hv * Wi2[k * 4 + 2];
      p3 += hv * Wi2[k * 4 + 3];
    }
  }
  #pragma unroll
  for (int m = 1; m < 32; m <<= 1) {
    p0 += __shfl_xor(p0, m, 32);
    p1 += __shfl_xor(p1, m, 32);
    p2 += __shfl_xor(p2, m, 32);
    p3 += __shfl_xor(p3, m, 32);
  }
  if (lane == 0) {
    float l0 = p0 + bi2[0], l1 = p1 + bi2[1], l2 = p2 + bi2[2], l3 = p3 + bi2[3];
    float mx = fmaxf(fmaxf(l0, l1), fmaxf(l2, l3));
    float e0 = expf(l0 - mx), e1 = expf(l1 - mx), e2 = expf(l2 - mx), e3 = expf(l3 - mx);
    float inv = 1.f / (e0 + e1 + e2 + e3);
    float* o = imp + (size_t)row * 4;
    o[0] = e0 * inv; o[1] = e1 * inv; o[2] = e2 * inv; o[3] = e3 * inv;
  }
}

// ---------------------------------------------------------------------------
// Host: workspace carving + launch sequence (all on `stream`, graph-safe).
// ---------------------------------------------------------------------------
static inline unsigned gblk(long total) { return (unsigned)((total + 255) / 256); }

extern "C" void kernel_launch(void* const* d_in, const int* in_sizes, int n_in,
                              void* d_out, int out_size, void* d_ws, size_t ws_size,
                              hipStream_t stream) {
  (void)in_sizes; (void)n_in; (void)out_size; (void)ws_size;

  const float* Wp[4]  = {(const float*)d_in[4],  (const float*)d_in[8],
                         (const float*)d_in[12], (const float*)d_in[16]};
  const float* bp[4]  = {(const float*)d_in[5],  (const float*)d_in[9],
                         (const float*)d_in[13], (const float*)d_in[17]};
  const float* gp[4]  = {(const float*)d_in[6],  (const float*)d_in[10],
                         (const float*)d_in[14], (const float*)d_in[18]};
  const float* bnp[4] = {(const float*)d_in[7],  (const float*)d_in[11],
                         (const float*)d_in[15], (const float*)d_in[19]};
  const float* Wv  = (const float*)d_in[20];
  const float* bv  = (const float*)d_in[21];
  const float* Wo  = (const float*)d_in[22];
  const float* bo  = (const float*)d_in[23];
  const float* Wi1 = (const float*)d_in[24];
  const float* bi1 = (const float*)d_in[25];
  const float* Wi2 = (const float*)d_in[26];
  const float* bi2 = (const float*)d_in[27];
  const float* Wg  = (const float*)d_in[28];
  const float* bg  = (const float*)d_in[29];
  const float* Wc  = (const float*)d_in[30];
  const float* bc  = (const float*)d_in[31];
  const float* Wout  = (const float*)d_in[32];
  const float* bout  = (const float*)d_in[33];
  const float* gout  = (const float*)d_in[34];
  const float* bnout = (const float*)d_in[35];

  const float* x_in[4] = {(const float*)d_in[0], (const float*)d_in[1],
                          (const float*)d_in[2], (const float*)d_in[3]};
  const int    din[4]  = {256, 768, 128, 256};

  // d_out layout: out[B,512] | att[4,B,512] | importance[B,4]
  float* out_out = (float*)d_out;
  float* out_att = out_out + BH;
  float* out_imp = out_att + 4 * BH;

  // workspace carving (f16 elements)
  half_t* ws = (half_t*)d_ws;
  size_t off = 0;
  size_t wptOff[4]; const size_t wptSz[4] = {512*256, 512*768, 512*128, 512*256};
  for (int m = 0; m < 4; ++m) { wptOff[m] = off; off += wptSz[m]; }
  half_t* WpT   = ws;
  half_t* Wv16  = ws + off; off += (size_t)12 * HDIM * HDIM;
  half_t* WoT   = ws + off; off += (size_t)12 * HDIM * HDIM;
  half_t* WeffT = ws + off; off += (size_t)12 * HDIM * HDIM;
  half_t* Wi1T  = ws + off; off += (size_t)HDIM * 2048;
  half_t* WgT   = ws + off; off += (size_t)HDIM * 2048;
  half_t* WcT   = ws + off; off += (size_t)HDIM * 2048;
  half_t* WoutT = ws + off; off += (size_t)HDIM * HDIM;
  half_t* P16   = ws + off; off += 4 * BH;
  half_t* att16 = ws + off; off += 4 * BH;
  half_t* fus16 = ws + off; off += BH;
  half_t* hid16 = ws + off; off += BH;
  float*  bsum  = (float*)(ws + off);   // 4*512 f32

  // --- 1. weight transpose/convert (f32 -> f16) ---
  for (int m = 0; m < 4; ++m) {
    long tot = (long)wptSz[m];
    k_transpose_f16<<<gblk(tot), 256, 0, stream>>>(Wp[m], WpT + wptOff[m], din[m], HDIM, tot);
  }
  {
    long tot = (long)12 * HDIM * HDIM;
    k_convert_f16  <<<gblk(tot), 256, 0, stream>>>(Wv, Wv16, tot);
    k_transpose_f16<<<gblk(tot), 256, 0, stream>>>(Wo, WoT, HDIM, HDIM, tot);
  }
  {
    long tot = (long)2048 * HDIM;
    k_transpose_f16<<<gblk(tot), 256, 0, stream>>>(Wi1, Wi1T, 2048, HDIM, tot);
    k_transpose_f16<<<gblk(tot), 256, 0, stream>>>(Wg,  WgT,  2048, HDIM, tot);
    k_transpose_f16<<<gblk(tot), 256, 0, stream>>>(Wc,  WcT,  2048, HDIM, tot);
  }
  {
    long tot = (long)HDIM * HDIM;
    k_transpose_f16<<<gblk(tot), 256, 0, stream>>>(Wout, WoutT, HDIM, HDIM, tot);
  }

  // --- 2. folded attention biases + combined weights WeffT = Wo^T @ Wv^T ---
  k_bsum<<<gblk(4 * HDIM), 256, 0, stream>>>(bv, Wo, bo, bsum);
  k_gemm_ff16<<<dim3(4, 4, 12), 256, 0, stream>>>(WoT, Wv16, WeffT);

  // --- 3. projections: P[m] = relu(LN(x@Wp + b)) ---
  for (int m = 0; m < 4; ++m)
    k_gemm_ln<float, true><<<BATCH / 64, 256, 0, stream>>>(
        x_in[m], din[m], WpT + wptOff[m], bp[m], gp[m], bnp[m],
        P16 + (size_t)m * BH, nullptr);

  // --- 4. attention collapse -> att (f32 to d_out, f16 scratch) ---
  k_attn<<<dim3(BATCH / 128, HDIM / 128, 4), 256, 0, stream>>>(
      P16, WeffT, bsum, out_att, att16);

  // --- 5. heads over cat (modality-remapped K=2048) ---
  k_cat_relu<<<dim3(BATCH / 128, HDIM / 128), 256, 0, stream>>>(att16, Wi1T, bi1, hid16);
  k_cat_gate<<<dim3(BATCH / 128, HDIM / 64), 256, 0, stream>>>(att16, WgT, WcT, bg, bc, fus16);

  // --- 6. out = LN(fused @ Wout + bout) ---
  k_gemm_ln<half_t, false><<<BATCH / 64, 256, 0, stream>>>(
      fus16, HDIM, WoutT, bout, gout, bnout, nullptr, out_out);

  // --- 7. importance softmax ---
  k_importance<<<BATCH / 8, 256, 0, stream>>>(hid16, Wi2, bi2, out_imp);
}
